// ScaledDotProductAttention_32066225832688
// MI455X (gfx1250) — matrix operand order repaired and checked by reading, not compiled, not run on hardware
//
#include <hip/hip_runtime.h>
#include <hip/hip_bf16.h>
#include <math.h>
#include <stdint.h>

typedef __attribute__((ext_vector_type(16))) __bf16 v16bf;
typedef __attribute__((ext_vector_type(8)))  __bf16 v8bf;
typedef __attribute__((ext_vector_type(4)))  __bf16 v4bf;
typedef __attribute__((ext_vector_type(8)))  float  v8f;

#define SDIM 2048
#define DDIM 64
#define BT   64     // q-tile rows == k-tile rows
#define LDP  80     // padded LDS row length (bf16 elems), keeps rows 32B-aligned

// Async global->LDS copy of 16B per lane (ASYNCcnt-tracked; IOFFSET applies to
// both the LDS and the global address, ISA §15.18 / §08 async).
#define ASYNC_CP16(LOFF, GPTR, OFF)                                        \
  asm volatile("global_load_async_to_lds_b128 %0, %1, off offset:" #OFF    \
               :: "v"(LOFF), "v"(GPTR) : "memory")

// Gather a B-operand column from LDS (rows k = base+s, base+16+s; stride LDP).
static __device__ inline v16bf ldB(const __bf16* p) {
  v16bf r;
#pragma unroll
  for (int i = 0; i < 8; ++i) { r[i] = p[i * LDP]; r[8 + i] = p[(16 + i) * LDP]; }
  return r;
}

__global__ __launch_bounds__(128) void fattn_fwd_kernel(
    const float* __restrict__ Q, const float* __restrict__ K,
    const float* __restrict__ V, const unsigned char* __restrict__ mask,
    float* __restrict__ O)
{
  // Raw fp32 staging (async-copy destination; read only in convert phase).
  __shared__ float  rawK[BT][DDIM];
  __shared__ float  rawV[BT][DDIM];
  // bf16 operand tiles.
  __shared__ __bf16 lK[DDIM][LDP];   // K^T tile: [d][k]  (B-operand for Q*K^T)
  __shared__ __bf16 lV[BT][LDP];     // V tile:   [k][d]  (B-operand for P*V)
  __shared__ __bf16 lP[4][16][LDP];  // per-wave P staging (S-frag -> A-frag)

  const int tid  = threadIdx.x;
  const int lane = tid & 31;
  const int wave = tid >> 5;
  const int l16  = lane & 15;
  const int half = lane >> 4;

  const int b     = blockIdx.x >> 5;   // 32 q-tiles per batch (2048/64)
  const int qTile = blockIdx.x & 31;
  const int q0    = qTile * BT;

  // Per-thread staging slice: one 32-float chunk of one row, 8 x 16B pieces.
  const int r  = tid >> 1;
  const int dh = (tid & 1) * 32;
  const uint32_t ldsK = (uint32_t)(uintptr_t)&rawK[r][dh];  // low 32 bits == LDS addr
  const uint32_t ldsV = (uint32_t)(uintptr_t)&rawV[r][dh];

  auto issue_tile = [&](int j) {
    const float* kp = K + ((size_t)b * SDIM + (size_t)j * BT + r) * DDIM + dh;
    const float* vp = V + ((size_t)b * SDIM + (size_t)j * BT + r) * DDIM + dh;
    ASYNC_CP16(ldsK, kp, 0);   ASYNC_CP16(ldsK, kp, 16);
    ASYNC_CP16(ldsK, kp, 32);  ASYNC_CP16(ldsK, kp, 48);
    ASYNC_CP16(ldsK, kp, 64);  ASYNC_CP16(ldsK, kp, 80);
    ASYNC_CP16(ldsK, kp, 96);  ASYNC_CP16(ldsK, kp, 112);
    ASYNC_CP16(ldsV, vp, 0);   ASYNC_CP16(ldsV, vp, 16);
    ASYNC_CP16(ldsV, vp, 32);  ASYNC_CP16(ldsV, vp, 48);
    ASYNC_CP16(ldsV, vp, 64);  ASYNC_CP16(ldsV, vp, 80);
    ASYNC_CP16(ldsV, vp, 96);  ASYNC_CP16(ldsV, vp, 112);
  };

  issue_tile(0);  // prime the pipe

  // ---- Q -> A-fragments (2 chunks of K-dim 32), pre-scaled by 1/sqrt(D) ----
  const float scale = 0.125f;
  const int   qRow  = q0 + wave * 16 + l16;
  const float* qp   = Q + ((size_t)b * SDIM + qRow) * DDIM;
  v16bf qa[2];
#pragma unroll
  for (int c = 0; c < 2; ++c) {
    const int cb = c * 32 + half * 8;
#pragma unroll
    for (int e = 0; e < 8; ++e) {
      qa[c][e]     = (__bf16)(qp[cb + e]      * scale);
      qa[c][8 + e] = (__bf16)(qp[cb + 16 + e] * scale);
    }
  }

  // ---- accumulators (C-frag: VGPR v <-> row half*8+v, column l16) ----
  v8f   oacc[4];
  float mrow[8], lrow[8];
#pragma unroll
  for (int nb = 0; nb < 4; ++nb)
#pragma unroll
    for (int v = 0; v < 8; ++v) oacc[nb][v] = 0.0f;
#pragma unroll
  for (int v = 0; v < 8; ++v) { mrow[v] = -3.0e38f; lrow[v] = 0.0f; }

  for (int j = 0; j <= qTile; ++j) {
    // ---- tile j raw data landed? (own copies, then workgroup-wide) ----
    asm volatile("s_wait_asynccnt 0" ::: "memory");
    __syncthreads();

    // ---- convert raw fp32 -> bf16 operand tiles (K transposed) ----
#pragma unroll
    for (int i = 0; i < 8; ++i) {
      const float4 kf = *(const float4*)&rawK[r][dh + i * 4];
      lK[dh + i * 4 + 0][r] = (__bf16)kf.x;
      lK[dh + i * 4 + 1][r] = (__bf16)kf.y;
      lK[dh + i * 4 + 2][r] = (__bf16)kf.z;
      lK[dh + i * 4 + 3][r] = (__bf16)kf.w;
      const float4 vf = *(const float4*)&rawV[r][dh + i * 4];
      v4bf vb;
      vb[0] = (__bf16)vf.x; vb[1] = (__bf16)vf.y;
      vb[2] = (__bf16)vf.z; vb[3] = (__bf16)vf.w;
      *(v4bf*)&lV[r][dh + i * 4] = vb;
    }
    __syncthreads();  // raw reads retired (dscnt), bf16 tiles visible

    // ---- overlap: fetch tile j+1 during this tile's compute ----
    if (j < qTile) issue_tile(j + 1);
    if (j + 2 <= qTile) {  // warm L2 two tiles ahead
      const size_t nbase = ((size_t)b * SDIM + (size_t)(j + 2) * BT) * DDIM;
      __builtin_prefetch(K + nbase + (size_t)tid * 32, 0, 0);
      __builtin_prefetch(V + nbase + (size_t)tid * 32, 0, 0);
    }

    // ---- S = (Q/sqrt(d)) * K^T : 4 column blocks x (2 WMMAs over K-dim 64) ----
    v8f s[4];
#pragma unroll
    for (int nb = 0; nb < 4; ++nb) {
      v8f z;
#pragma unroll
      for (int v = 0; v < 8; ++v) z[v] = 0.0f;
      const v16bf b0 = ldB(&lK[half * 8][nb * 16 + l16]);        // d rows 0..31
      s[nb] = __builtin_amdgcn_wmma_f32_16x16x32_bf16(
          false, qa[0], false, b0, (short)0, z, false, false);
      const v16bf b1 = ldB(&lK[32 + half * 8][nb * 16 + l16]);   // d rows 32..63
      s[nb] = __builtin_amdgcn_wmma_f32_16x16x32_bf16(
          false, qa[1], false, b1, (short)0, s[nb], false, false);
    }

    // ---- padding mask (-10000 per masked k column) ----
#pragma unroll
    for (int nb = 0; nb < 4; ++nb) {
      const int   kcol = j * BT + nb * 16 + l16;
      const float pen  = mask[(size_t)b * SDIM + kcol] ? 0.0f : 10000.0f;
#pragma unroll
      for (int v = 0; v < 8; ++v) s[nb][v] -= pen;
    }
    // ---- causal mask: only the diagonal tile needs per-element work ----
    if (j == qTile) {
#pragma unroll
      for (int nb = 0; nb < 4; ++nb) {
        const int kcol = j * BT + nb * 16 + l16;
#pragma unroll
        for (int v = 0; v < 8; ++v) {
          const int qr = q0 + wave * 16 + half * 8 + v;
          if (kcol > qr) s[nb][v] -= 10000.0f;
        }
      }
    }

    // ---- online softmax (row stats via 16-lane xor reductions) ----
#pragma unroll
    for (int v = 0; v < 8; ++v) {
      float tm = fmaxf(fmaxf(s[0][v], s[1][v]), fmaxf(s[2][v], s[3][v]));
#pragma unroll
      for (int off = 8; off >= 1; off >>= 1)
        tm = fmaxf(tm, __shfl_xor(tm, off, 32));
      const float mNew  = fmaxf(mrow[v], tm);
      const float alpha = __expf(mrow[v] - mNew);
      mrow[v] = mNew;
      float ps = 0.0f;
#pragma unroll
      for (int nb = 0; nb < 4; ++nb) {
        const float p = __expf(s[nb][v] - mNew);
        s[nb][v] = p;
        ps += p;
      }
#pragma unroll
      for (int off = 8; off >= 1; off >>= 1)
        ps += __shfl_xor(ps, off, 32);
      lrow[v] = lrow[v] * alpha + ps;
#pragma unroll
      for (int nb = 0; nb < 4; ++nb) oacc[nb][v] *= alpha;
    }

    // ---- P: S-frag -> LDS -> A-frag (same-wave LDS is in-order, ISA §7.3) ----
#pragma unroll
    for (int nb = 0; nb < 4; ++nb)
#pragma unroll
      for (int v = 0; v < 8; ++v)
        lP[wave][half * 8 + v][nb * 16 + l16] = (__bf16)s[nb][v];

    v16bf pa[2];
#pragma unroll
    for (int c = 0; c < 2; ++c) {
      const int cb = c * 32 + half * 8;
      const v8bf lo = *(const v8bf*)&lP[wave][l16][cb];
      const v8bf hi = *(const v8bf*)&lP[wave][l16][cb + 16];
#pragma unroll
      for (int e = 0; e < 8; ++e) { pa[c][e] = lo[e]; pa[c][8 + e] = hi[e]; }
    }

    // ---- O += P * V : 4 d-blocks x (2 WMMAs over k-dim 64) ----
#pragma unroll
    for (int nb = 0; nb < 4; ++nb) {
      const v16bf b0 = ldB(&lV[half * 8][nb * 16 + l16]);        // k rows 0..31
      oacc[nb] = __builtin_amdgcn_wmma_f32_16x16x32_bf16(
          false, pa[0], false, b0, (short)0, oacc[nb], false, false);
      const v16bf b1 = ldB(&lV[32 + half * 8][nb * 16 + l16]);   // k rows 32..63
      oacc[nb] = __builtin_amdgcn_wmma_f32_16x16x32_bf16(
          false, pa[1], false, b1, (short)0, oacc[nb], false, false);
    }
    // no trailing barrier: next iteration's wait+barrier fences lK/lV reuse
  }

  // ---- epilogue: O / rowsum, coalesced fp32 stores ----
  float* op = O + ((size_t)b * SDIM + q0 + wave * 16 + half * 8) * DDIM;
#pragma unroll
  for (int v = 0; v < 8; ++v) {
    const float inv = __builtin_amdgcn_rcpf(lrow[v]);
#pragma unroll
    for (int nb = 0; nb < 4; ++nb)
      op[(size_t)v * DDIM + nb * 16 + l16] = oacc[nb][v] * inv;
  }
}

extern "C" void kernel_launch(void* const* d_in, const int* in_sizes, int n_in,
                              void* d_out, int out_size, void* d_ws, size_t ws_size,
                              hipStream_t stream) {
  const float*         Q    = (const float*)d_in[0];
  const float*         K    = (const float*)d_in[1];
  const float*         V    = (const float*)d_in[2];
  const unsigned char* mask = (const unsigned char*)d_in[3];
  float*               O    = (float*)d_out;

  const int B = 8, S = SDIM;
  dim3 grid(B * (S / BT));   // 8 * 32 = 256 workgroups
  dim3 block(128);           // 4 waves (wave32)
  fattn_fwd_kernel<<<grid, block, 0, stream>>>(Q, K, V, mask, O);
}
